// EdgeMLP_13116830122419
// MI455X (gfx1250) — matrix operand order repaired and checked by reading, not compiled, not run on hardware
//
#include <hip/hip_runtime.h>

typedef __attribute__((ext_vector_type(16))) _Float16 v16h;
typedef __attribute__((ext_vector_type(8)))  _Float16 v8h;
typedef __attribute__((ext_vector_type(4)))  _Float16 v4h;
typedef __attribute__((ext_vector_type(8)))  float    v8f;

#define DFEAT 128          // node/edge feature dim == DOUT
#define DIN   384          // 3 * 128 concat
#define TILE_M 16          // edges per WMMA tile
#define TILES_PER_BLOCK 16
#define EDGES_PER_BLOCK (TILE_M * TILES_PER_BLOCK)   // 256
#define KSTEPS (DIN / 32)  // 12 WMMA k-steps
#define HSTRIDE 392        // halves per H row (384 + 8 pad, keeps 16B align)
#define YSTRIDE 132        // floats per Y row (128 + 4 pad -> conflict-free halves)

__global__ __launch_bounds__(256)
void edge_mlp_ln_kernel(const float* __restrict__ node_x,
                        const float* __restrict__ edge_feat,
                        const int*   __restrict__ edge_index,   // (2,E) int32
                        const float* __restrict__ W,            // (384,128) row-major
                        const float* __restrict__ bias,         // (128)
                        const float* __restrict__ ln_gamma,     // (128)
                        const float* __restrict__ ln_beta,      // (128)
                        float*       __restrict__ out,          // (E,128)
                        int E)
{
    __shared__ _Float16 hTile[TILE_M * HSTRIDE];   // 12.25 KB: gathered A tile (f16)
    __shared__ float    yTile[TILE_M * YSTRIDE];   //  8.25 KB: pre-LN outputs (f32)

    const int tid    = threadIdx.x;
    const int lane   = tid & 31;
    const int wave   = tid >> 5;          // 0..7 -> 16-column slab of the output
    const int laneLo = lane & 15;
    const int laneHi = lane >> 4;

    // ---------------------------------------------------------------
    // Preload this wave's B fragments (W columns [16*wave, 16*wave+16))
    // into registers, converted f32 -> f16. Layout per ISA: B 32x16 f16,
    // elems 0-7 hold K=8*laneHi+e, elems 8-15 hold K=16+8*laneHi+(e-8) (cols = lane).
    // 12 fragments * 8 VGPRs = 96 VGPRs, reused for all 16 edge tiles.
    // ---------------------------------------------------------------
    const int nCol = wave * 16 + laneLo;
    v16h bfrag[KSTEPS];
#pragma unroll
    for (int t = 0; t < KSTEPS; ++t) {
        v16h bv;
#pragma unroll
        for (int e = 0; e < 16; ++e) {
            bv[e] = (_Float16)W[(t * 32 + laneHi * 8 + (e & 7) + (e >> 3) * 16) * DFEAT + nCol];
        }
        bfrag[t] = bv;
    }
    const float bval = bias[nCol];

    const long long blockEdge0 = (long long)blockIdx.x * EDGES_PER_BLOCK;

    for (int tile = 0; tile < TILES_PER_BLOCK; ++tile) {
        const long long eBase = blockEdge0 + (long long)tile * TILE_M;
        if (eBase >= (long long)E) break;   // uniform across block

        // -----------------------------------------------------------
        // Gather H tile: 16 rows x 384 cols = [x_i | ef | x_j] -> f16 LDS.
        // 1536 float4 segments, 6 per thread. Node gathers hit L2.
        // -----------------------------------------------------------
#pragma unroll
        for (int q = 0; q < 6; ++q) {
            const int linear = tid + q * 256;     // 0..1535
            const int row    = linear / 96;       // 96 float4 per row
            const int seg    = linear % 96;
            const long long e = eBase + row;
            if (e < (long long)E) {
                const float* src;
                if (seg < 32) {
                    const int idx = edge_index[e];                     // src node
                    src = node_x + (long long)idx * DFEAT + seg * 4;
                } else if (seg < 64) {
                    src = edge_feat + e * DFEAT + (seg - 32) * 4;      // edge feat
                } else {
                    const int idx = edge_index[(long long)E + e];      // dst node
                    src = node_x + (long long)idx * DFEAT + (seg - 64) * 4;
                }
                const float4 v = *(const float4*)src;
                v4h pk = { (_Float16)v.x, (_Float16)v.y,
                           (_Float16)v.z, (_Float16)v.w };
                *(v4h*)&hTile[row * HSTRIDE + seg * 4] = pk;           // ds_store_b64
            }
        }
        // prefetch next tile's streaming edge features while we compute
        if (eBase + TILE_M < (long long)E)
            __builtin_prefetch(edge_feat + (eBase + TILE_M) * DFEAT + lane, 0, 1);
        __syncthreads();

        // -----------------------------------------------------------
        // WMMA: acc(16x16) = H(16x384) x Wslab(384x16), f32 accumulate.
        // A fragment per ISA: lanes 0-15 = rows M; VGPR0-3 hold K-octet
        // at 32t + 8*laneHi, VGPR4-7 at 32t + 16 + 8*laneHi -> 2x b128.
        // -----------------------------------------------------------
        v8f acc = {};
        const _Float16* arow = &hTile[laneLo * HSTRIDE];
#pragma unroll
        for (int t = 0; t < KSTEPS; ++t) {
            const v8h lo = *(const v8h*)(arow + t * 32 + laneHi * 8);       // ds_read_b128
            const v8h hi = *(const v8h*)(arow + t * 32 + 16 + laneHi * 8);  // ds_read_b128
            v16h av;
#pragma unroll
            for (int i = 0; i < 8; ++i) { av[i] = lo[i]; av[8 + i] = hi[i]; }
            acc = __builtin_amdgcn_wmma_f32_16x16x32_f16(
                      false, av, false, bfrag[t], (short)0, acc, false, false);
        }

        // stage biased result: C/D layout -> row v (+8 for hi lanes), col nCol
#pragma unroll
        for (int v = 0; v < 8; ++v) {
            yTile[(v + laneHi * 8) * YSTRIDE + nCol] = acc[v] + bval;
        }
        __syncthreads();

        // -----------------------------------------------------------
        // LayerNorm over the 128-dim row; wave w handles rows 2w, 2w+1.
        // Wave32 shfl_xor tree reduction, float4 coalesced store.
        // -----------------------------------------------------------
#pragma unroll
        for (int rr = 0; rr < 2; ++rr) {
            const int row = wave * 2 + rr;
            const long long e = eBase + row;
            const float4 y = *(const float4*)&yTile[row * YSTRIDE + lane * 4];
            float s  = y.x + y.y + y.z + y.w;
            float ss = y.x * y.x + y.y * y.y + y.z * y.z + y.w * y.w;
#pragma unroll
            for (int off = 16; off >= 1; off >>= 1) {
                s  += __shfl_xor(s,  off, 32);
                ss += __shfl_xor(ss, off, 32);
            }
            const float mu   = s * (1.0f / 128.0f);
            const float var  = ss * (1.0f / 128.0f) - mu * mu;
            const float rstd = rsqrtf(var + 1e-5f);
            if (e < (long long)E) {
                const float4 g  = *(const float4*)(ln_gamma + lane * 4);
                const float4 be = *(const float4*)(ln_beta  + lane * 4);
                float4 o;
                o.x = (y.x - mu) * rstd * g.x + be.x;
                o.y = (y.y - mu) * rstd * g.y + be.y;
                o.z = (y.z - mu) * rstd * g.z + be.z;
                o.w = (y.w - mu) * rstd * g.w + be.w;
                *(float4*)(out + e * (long long)DFEAT + lane * 4) = o;  // b128 store
            }
        }
        __syncthreads();
    }
}

extern "C" void kernel_launch(void* const* d_in, const int* in_sizes, int n_in,
                              void* d_out, int out_size, void* d_ws, size_t ws_size,
                              hipStream_t stream)
{
    const float* node_x = (const float*)d_in[0];
    const float* ef0    = (const float*)d_in[1];
    const float* ef1    = (const float*)d_in[2];
    const int*   ei0    = (const int*)  d_in[3];   // JAX default demotes int64->int32
    const int*   ei1    = (const int*)  d_in[4];
    const float* W0     = (const float*)d_in[5];
    const float* b0     = (const float*)d_in[6];
    const float* W1     = (const float*)d_in[7];
    const float* b1     = (const float*)d_in[8];
    const float* gamma_ = (const float*)d_in[9];
    const float* beta_  = (const float*)d_in[10];

    const int E = in_sizes[1] / DFEAT;             // edge_feat0 = E*128
    float* out0 = (float*)d_out;
    float* out1 = out0 + (size_t)E * DFEAT;

    const int nBlocks = (E + EDGES_PER_BLOCK - 1) / EDGES_PER_BLOCK;
    dim3 grid(nBlocks), block(256);

    edge_mlp_ln_kernel<<<grid, block, 0, stream>>>(
        node_x, ef0, ei0, W0, b0, gamma_, beta_, out0, E);
    edge_mlp_ln_kernel<<<grid, block, 0, stream>>>(
        node_x, ef1, ei1, W1, b1, gamma_, beta_, out1, E);
}
